// SubstrateGNN_69853348102175
// MI455X (gfx1250) — matrix operand order repaired and checked by reading, compiled, not run on hardware
//
#include <hip/hip_runtime.h>
#include <hip/hip_bf16.h>

// ---------------------------------------------------------------------------
// GATv2 3-layer GNN for MI455X (gfx1250, wave32, WMMA)
// GEMMs: f16 inputs, f32 accumulate via v_wmma_f32_16x16x32_f16.
// Each wave computes 4 column tiles (register-level A reuse: 1 A fragment,
// 4 B fragments, 4 accumulators) -> 4x less L2 traffic on activations.
// Edge attention: float4 vmem + f32 atomics (node tables are L2-resident).
// ---------------------------------------------------------------------------

typedef __attribute__((ext_vector_type(16))) _Float16 v16h;
typedef __attribute__((ext_vector_type(8)))  float    v8f;

union AFrag { v16h v; uint4 q[2]; };

#define NEG_SLOPE 0.2f
#define CT 4   // column tiles per wave

// ----------------------------- utility kernels -----------------------------

__global__ void fill_f32(float* __restrict__ p, float v, int n) {
    int i = blockIdx.x * blockDim.x + threadIdx.x;
    if (i < n) p[i] = v;
}

__global__ void cast_f16_kernel(const float* __restrict__ src,
                                _Float16* __restrict__ dst, int n) {
    int i = blockIdx.x * blockDim.x + threadIdx.x;
    if (i < n) dst[i] = (_Float16)src[i];
}

// W [K,M] row-major f32  ->  Wt [M,K] row-major f16
__global__ void transpose_cast_kernel(const float* __restrict__ W,
                                      _Float16* __restrict__ Wt, int K, int M) {
    int i = blockIdx.x * blockDim.x + threadIdx.x;
    if (i < K * M) {
        int k = i / M, m = i % M;
        Wt[(size_t)m * K + k] = (_Float16)W[i];
    }
}

// ------------------------------- WMMA GEMM ---------------------------------
// out[N,M] = A16[N,K] * Wt[M,K]^T + bias[M]   (f16 in, f32 out)
// block = (32, M/64): each wave owns one 16-row stripe x 64 output columns
// (4 WMMA col tiles). grid = (1, N/16). N/16 must be exact (50000/16=3125).
__global__ __launch_bounds__(128)
void wmma_gemm_bias(const _Float16* __restrict__ A,
                    const _Float16* __restrict__ Wt,
                    const float* __restrict__ bias,
                    float* __restrict__ out,
                    int N, int K, int M) {
    const int lane    = threadIdx.x;          // 0..31
    const int wave    = threadIdx.y;          // 0..(M/64-1)
    const int rowTile = blockIdx.y;
    const int l16     = lane & 15;
    const int h       = lane >> 4;            // lane half: selects K subrange

    const int row     = rowTile * 16 + l16;   // A fragment: M-dim = lane%16
    const int colBase = wave * 64 + l16;      // first of 4 col tiles

    AFrag a, b[CT];
    v8f c[CT] = {};

    const _Float16* arow = A + (size_t)row * K;
    const _Float16* brow[CT];
    #pragma unroll
    for (int j = 0; j < CT; ++j)
        brow[j] = Wt + (size_t)(colBase + j * 16) * K;

    for (int k0 = 0; k0 < K; k0 += 32) {
        // A 16x32: lane half h holds K = k0+h*8+0..7 and k0+16+h*8+0..7
        const _Float16* pa = arow + k0 + h * 8;
        a.q[0] = *(const uint4*)(pa);
        a.q[1] = *(const uint4*)(pa + 16);
        // B 32x16 per tile: lane half h holds K = k0+h*8+0..7 and k0+16+h*8+0..7
        #pragma unroll
        for (int j = 0; j < CT; ++j) {
            const _Float16* pb = brow[j] + k0 + h * 8;
            b[j].q[0] = *(const uint4*)(pb);
            b[j].q[1] = *(const uint4*)(pb + 16);
        }
        #pragma unroll
        for (int j = 0; j < CT; ++j)
            c[j] = __builtin_amdgcn_wmma_f32_16x16x32_f16(
                       false, a.v, false, b[j].v, (short)0, c[j], false, false);
    }

    // D 16x16 f32: VGPR v, lanes 0-15 -> row rowTile*16+v, lanes 16-31 -> +8
    #pragma unroll
    for (int j = 0; j < CT; ++j) {
        int col = colBase + j * 16;
        float bv = bias[col];
        #pragma unroll
        for (int v = 0; v < 8; ++v) {
            int r = rowTile * 16 + v + h * 8;
            out[(size_t)r * M + col] = c[j][v] + bv;
        }
    }
}

// --------------------------- edge attention kernels ------------------------

__device__ __forceinline__ void atomicMaxFloat(float* addr, float val) {
    if (val >= 0.f) atomicMax((int*)addr, __float_as_int(val));
    else            atomicMin((unsigned int*)addr, __float_as_uint(val));
}

// Pass 1: logit[e,h] = att[h] . leaky_relu(xl[src] + xr[dst]);  seg-max(dst)
__global__ void attn_logits(const float* __restrict__ xl,
                            const float* __restrict__ xr,
                            const float* __restrict__ att,
                            const int* __restrict__ ei,
                            int E, int E2, int H, int C,
                            float* __restrict__ logit,
                            float* __restrict__ maxb) {
    int t = blockIdx.x * blockDim.x + threadIdx.x;
    if (t >= E2 * H) return;
    int hh = t % H, e = t / H;
    int s = (e < E) ? ei[e]     : (e - E);
    int d = (e < E) ? ei[E + e] : (e - E);
    int HC = H * C;
    const float4* pl = (const float4*)(xl + (size_t)s * HC + hh * C);
    const float4* pr = (const float4*)(xr + (size_t)d * HC + hh * C);
    const float4* pa = (const float4*)(att + hh * C);
    float acc = 0.f;
    for (int i = 0; i < (C >> 2); ++i) {
        float4 a = pl[i], b = pr[i], w = pa[i];
        float v;
        v = a.x + b.x; v = v > 0.f ? v : NEG_SLOPE * v; acc += v * w.x;
        v = a.y + b.y; v = v > 0.f ? v : NEG_SLOPE * v; acc += v * w.y;
        v = a.z + b.z; v = v > 0.f ? v : NEG_SLOPE * v; acc += v * w.z;
        v = a.w + b.w; v = v > 0.f ? v : NEG_SLOPE * v; acc += v * w.w;
    }
    logit[t] = acc;
    atomicMaxFloat(maxb + (size_t)d * H + hh, acc);
}

// Pass 2: ex = exp(logit - max[dst]);  seg-sum(dst)
__global__ void edge_expsum(const int* __restrict__ ei,
                            int E, int E2, int H,
                            float* __restrict__ logit,
                            const float* __restrict__ maxb,
                            float* __restrict__ sumb) {
    int t = blockIdx.x * blockDim.x + threadIdx.x;
    if (t >= E2 * H) return;
    int hh = t % H, e = t / H;
    int d = (e < E) ? ei[E + e] : (e - E);
    float ex = expf(logit[t] - maxb[(size_t)d * H + hh]);
    logit[t] = ex;
    atomicAdd(sumb + (size_t)d * H + hh, ex);
}

// Pass 3: out[dst] += (ex/denom[dst]) * xl[src]   (float4 chunks, f32 atomics)
__global__ void edge_aggregate(const int* __restrict__ ei,
                               int E, int E2, int H, int C,
                               const float* __restrict__ xl,
                               const float* __restrict__ logit,
                               const float* __restrict__ sumb,
                               float* __restrict__ out) {
    int t = blockIdx.x * blockDim.x + threadIdx.x;
    int CH4 = C >> 2;
    if (t >= E2 * H * CH4) return;
    int ci = t % CH4;
    int t2 = t / CH4;
    int hh = t2 % H, e = t2 / H;
    int s = (e < E) ? ei[e]     : (e - E);
    int d = (e < E) ? ei[E + e] : (e - E);
    float alpha = logit[t2] / sumb[(size_t)d * H + hh];
    int HC = H * C;
    const float4* ps = (const float4*)(xl + (size_t)s * HC + hh * C);
    float4 v = ps[ci];
    float* po = out + (size_t)d * HC + hh * C + ci * 4;
    atomicAdd(po + 0, alpha * v.x);
    atomicAdd(po + 1, alpha * v.y);
    atomicAdd(po + 2, alpha * v.z);
    atomicAdd(po + 3, alpha * v.w);
}

// out += bias ; optional ELU (alpha=1)
__global__ void bias_act(float* __restrict__ buf, const float* __restrict__ bias,
                         int n, int M, int do_elu) {
    int i = blockIdx.x * blockDim.x + threadIdx.x;
    if (i >= n) return;
    float v = buf[i] + bias[i % M];
    if (do_elu) v = v > 0.f ? v : (expf(v) - 1.f);
    buf[i] = v;
}

// ------------------------------ pooling ------------------------------------

__global__ void pool_count(const int* __restrict__ batch, float* __restrict__ cnt,
                           int N) {
    int i = blockIdx.x * blockDim.x + threadIdx.x;
    if (i < N) atomicAdd(cnt + batch[i], 1.f);
}

__global__ void pool_sum(const int* __restrict__ batch,
                         const float* __restrict__ h,
                         float* __restrict__ sums, int N, int D) {
    int t = blockIdx.x * blockDim.x + threadIdx.x;
    if (t >= N * D) return;
    int i = t / D, j = t % D;
    atomicAdd(sums + (size_t)batch[i] * D + j, h[t]);
}

__global__ void pool_div(float* __restrict__ o, const float* __restrict__ cnt,
                         int G, int D) {
    int t = blockIdx.x * blockDim.x + threadIdx.x;
    if (t < G * D) o[t] /= fmaxf(cnt[t / D], 1.f);
}

// ------------------------------- host side ---------------------------------

static inline int cdiv(int a, int b) { return (a + b - 1) / b; }

static void run_gatv2_layer(const float* act_in, int Kin,
                            const float* Wl, const float* bl,
                            const float* Wr, const float* br,
                            const float* att, const float* bout,
                            int H, int C, int do_elu,
                            _Float16* a16, _Float16* wlt, _Float16* wrt,
                            float* xl, float* xr, float* outb,
                            float* logitb, float* maxb, float* sumb,
                            const int* ei, int N, int E, hipStream_t st) {
    const int Mout = H * C;   // output width per node
    const int E2   = E + N;   // with self loops
    const int T    = 256;

    // cast activations + transpose weights to f16
    cast_f16_kernel<<<cdiv(N * Kin, T), T, 0, st>>>(act_in, a16, N * Kin);
    transpose_cast_kernel<<<cdiv(Kin * Mout, T), T, 0, st>>>(Wl, wlt, Kin, Mout);
    transpose_cast_kernel<<<cdiv(Kin * Mout, T), T, 0, st>>>(Wr, wrt, Kin, Mout);

    // two WMMA GEMMs: xl = a@Wl+bl, xr = a@Wr+br
    // block covers one 16-row stripe x all Mout columns (each wave: 64 cols)
    dim3 gb(32, Mout / 64, 1);
    dim3 gg(1, N / 16, 1);
    wmma_gemm_bias<<<gg, gb, 0, st>>>(a16, wlt, bl, xl, N, Kin, Mout);
    wmma_gemm_bias<<<gg, gb, 0, st>>>(a16, wrt, br, xr, N, Kin, Mout);

    // init segment buffers + output accumulator
    fill_f32<<<cdiv(N * H, T), T, 0, st>>>(maxb, -__builtin_inff(), N * H);
    fill_f32<<<cdiv(N * H, T), T, 0, st>>>(sumb, 0.f, N * H);
    fill_f32<<<cdiv(N * Mout, T), T, 0, st>>>(outb, 0.f, N * Mout);

    // edge attention
    attn_logits<<<cdiv(E2 * H, T), T, 0, st>>>(xl, xr, att, ei, E, E2, H, C,
                                               logitb, maxb);
    edge_expsum<<<cdiv(E2 * H, T), T, 0, st>>>(ei, E, E2, H, logitb, maxb, sumb);
    edge_aggregate<<<cdiv(E2 * H * (C / 4), T), T, 0, st>>>(
        ei, E, E2, H, C, xl, logitb, sumb, outb);

    // bias (+ ELU)
    bias_act<<<cdiv(N * Mout, T), T, 0, st>>>(outb, bout, N * Mout, Mout, do_elu);
}

extern "C" void kernel_launch(void* const* d_in, const int* in_sizes, int n_in,
                              void* d_out, int out_size, void* d_ws, size_t ws_size,
                              hipStream_t stream) {
    // inputs (setup_inputs order)
    const float* x     = (const float*)d_in[0];
    const int*   ei    = (const int*)  d_in[1];   // [2,E] row-major
    const int*   batch = (const int*)  d_in[2];
    const float* Wl1 = (const float*)d_in[3],  *bl1 = (const float*)d_in[4];
    const float* Wr1 = (const float*)d_in[5],  *br1 = (const float*)d_in[6];
    const float* at1 = (const float*)d_in[7],  *b1  = (const float*)d_in[8];
    const float* Wl2 = (const float*)d_in[9],  *bl2 = (const float*)d_in[10];
    const float* Wr2 = (const float*)d_in[11], *br2 = (const float*)d_in[12];
    const float* at2 = (const float*)d_in[13], *b2  = (const float*)d_in[14];
    const float* Wl3 = (const float*)d_in[15], *bl3 = (const float*)d_in[16];
    const float* Wr3 = (const float*)d_in[17], *br3 = (const float*)d_in[18];
    const float* at3 = (const float*)d_in[19], *b3  = (const float*)d_in[20];

    const int Din = 64, H = 4, C = 64, HC = 256, Dout = 128;
    const int N  = in_sizes[0] / Din;
    const int E  = in_sizes[1] / 2;
    const int E2 = E + N;
    const int G  = out_size / Dout;
    const int T  = 256;

    // ---- workspace carving (256B aligned) ----
    char* w = (char*)d_ws;
    auto carve = [&](size_t bytes) -> void* {
        void* p = (void*)w;
        w += (bytes + 255) & ~(size_t)255;
        return p;
    };
    _Float16* a16   = (_Float16*)carve((size_t)N * HC * sizeof(_Float16));
    _Float16* wlt   = (_Float16*)carve((size_t)HC * HC * sizeof(_Float16));
    _Float16* wrt   = (_Float16*)carve((size_t)HC * HC * sizeof(_Float16));
    float*    xl    = (float*)carve((size_t)N * HC * sizeof(float));
    float*    xr    = (float*)carve((size_t)N * HC * sizeof(float));
    float*    actb  = (float*)carve((size_t)N * HC * sizeof(float));  // out/act
    float*    logitb= (float*)carve((size_t)E2 * H * sizeof(float));
    float*    maxb  = (float*)carve((size_t)N * H * sizeof(float));
    float*    sumb  = (float*)carve((size_t)N * H * sizeof(float));
    float*    cnt   = (float*)carve((size_t)G * sizeof(float));
    (void)ws_size; (void)n_in;

    // layer 1: 64 -> 4x64 concat, ELU     (out -> actb)
    run_gatv2_layer(x, Din, Wl1, bl1, Wr1, br1, at1, b1, H, C, 1,
                    a16, wlt, wrt, xl, xr, actb, logitb, maxb, sumb,
                    ei, N, E, stream);
    // layer 2: 256 -> 4x64 concat, ELU    (actb cast to a16 first, then reused)
    run_gatv2_layer(actb, HC, Wl2, bl2, Wr2, br2, at2, b2, H, C, 1,
                    a16, wlt, wrt, xl, xr, actb, logitb, maxb, sumb,
                    ei, N, E, stream);
    // layer 3: 256 -> 128, 1 head, no ELU
    run_gatv2_layer(actb, HC, Wl3, bl3, Wr3, br3, at3, b3, 1, Dout, 0,
                    a16, wlt, wrt, xl, xr, actb, logitb, maxb, sumb,
                    ei, N, E, stream);

    // global mean pool
    float* out = (float*)d_out;
    fill_f32<<<cdiv(G * Dout, T), T, 0, stream>>>(out, 0.f, G * Dout);
    fill_f32<<<cdiv(G, T), T, 0, stream>>>(cnt, 0.f, G);
    pool_count<<<cdiv(N, T), T, 0, stream>>>(batch, cnt, N);
    pool_sum<<<cdiv(N * Dout, T), T, 0, stream>>>(batch, actb, out, N, Dout);
    pool_div<<<cdiv(G * Dout, T), T, 0, stream>>>(out, cnt, G, Dout);
}
